// MultiheadAttention_35751307771944
// MI455X (gfx1250) — matrix operand order repaired and checked by reading, not compiled, not run on hardware
//
#include <hip/hip_runtime.h>
#include <hip/hip_bf16.h>
#include <math.h>

typedef __attribute__((ext_vector_type(16))) __bf16 v16bf;
typedef __attribute__((ext_vector_type(8)))  __bf16 v8bf;
typedef __attribute__((ext_vector_type(8)))  float  v8f;
typedef int v4i_gcc __attribute__((vector_size(16)));   // matches builtin param

#define B_   4
#define L_   2048
#define VEC_ 512
#define H_   8
#define D_   64
#define EPS_ 1e-3f
#define NEG_BIG (-3.0e38f)

#if __has_builtin(__builtin_amdgcn_global_load_async_to_lds_b128)
#define ASYNC_LDS 1
#else
#define ASYNC_LDS 0
#endif

// 16-byte global -> LDS copy; async (ASYNCcnt-tracked, no VGPR round trip)
// when the toolchain exposes the gfx1250 builtin, else plain load+store.
__device__ __forceinline__ void cp16(const __bf16* g, __bf16* l) {
#if ASYNC_LDS
    __builtin_amdgcn_global_load_async_to_lds_b128(
        (__attribute__((address_space(1))) v4i_gcc*)g,
        (__attribute__((address_space(3))) v4i_gcc*)l, 0, 0);
#else
    *(uint4*)l = *(const uint4*)g;
#endif
}
__device__ __forceinline__ void wait_cp() {
#if ASYNC_LDS
    asm volatile("s_wait_asynccnt 0x0" ::: "memory");
#endif
}

// ---- fragment load helpers: 16B-aligned contiguous chunks -> b128 loads ----
__device__ __forceinline__ v16bf cat8(v8bf lo, v8bf hi) {
    return __builtin_shufflevector(lo, hi, 0,1,2,3,4,5,6,7,8,9,10,11,12,13,14,15);
}
// B fragment: lane%16=N, halves hold K 0-15/16-31 -> 32 contiguous bytes
__device__ __forceinline__ v16bf ld16(const __bf16* p) {
    return cat8(*(const v8bf*)p, *(const v8bf*)(p + 8));
}
// A fragment (16-bit 16x32): per lane-half one contiguous chunk
// at K = half*16 .. half*16+15 (matches B fragment order)
__device__ __forceinline__ v16bf ld_a(const __bf16* row, int half) {
    return cat8(*(const v8bf*)(row + half * 16), *(const v8bf*)(row + half * 16 + 8));
}

__device__ __forceinline__ float rmax16(float v) {
    v = fmaxf(v, __shfl_xor(v, 1, 32));
    v = fmaxf(v, __shfl_xor(v, 2, 32));
    v = fmaxf(v, __shfl_xor(v, 4, 32));
    v = fmaxf(v, __shfl_xor(v, 8, 32));
    return v;
}
__device__ __forceinline__ float rsum16(float v) {
    v += __shfl_xor(v, 1, 32);
    v += __shfl_xor(v, 2, 32);
    v += __shfl_xor(v, 4, 32);
    v += __shfl_xor(v, 8, 32);
    return v;
}

// ---------------------------------------------------------------------------
// Kernel 0a: f32 -> bf16 conversion of query/key/value (one pass; halves all
// downstream activation bytes through L2/HBM).
// ---------------------------------------------------------------------------
__global__ __launch_bounds__(256)
void cvt_bf16_kernel(const float* __restrict__ q, const float* __restrict__ k,
                     const float* __restrict__ v,
                     __bf16* __restrict__ qo, __bf16* __restrict__ ko,
                     __bf16* __restrict__ vo)
{
    const int z = blockIdx.y;
    const float* src = (z == 0) ? q : (z == 1) ? k : v;
    __bf16* dst      = (z == 0) ? qo : (z == 1) ? ko : vo;
    const size_t i = ((size_t)blockIdx.x * 256 + threadIdx.x) * 8;
    const float4 a = *(const float4*)(src + i);
    const float4 b = *(const float4*)(src + i + 4);
    union { uint4 u; __bf16 e[8]; } pk;
    pk.e[0] = (__bf16)a.x; pk.e[1] = (__bf16)a.y;
    pk.e[2] = (__bf16)a.z; pk.e[3] = (__bf16)a.w;
    pk.e[4] = (__bf16)b.x; pk.e[5] = (__bf16)b.y;
    pk.e[6] = (__bf16)b.z; pk.e[7] = (__bf16)b.w;
    *(uint4*)(dst + i) = pk.u;
}

// ---------------------------------------------------------------------------
// Kernel 0b: W[h][v][d] f32 -> Wt[h][d][v] bf16 (tiny one-time transpose so
// projection B-fragments are contiguous 32B LDS loads).
// ---------------------------------------------------------------------------
__global__ __launch_bounds__(256)
void wt_kernel(const float* __restrict__ Wq, const float* __restrict__ Wk,
               const float* __restrict__ Wv,
               __bf16* __restrict__ tq, __bf16* __restrict__ tk,
               __bf16* __restrict__ tv)
{
    const int z = blockIdx.y;
    const float* W = (z == 0) ? Wq : (z == 1) ? Wk : Wv;
    __bf16* T      = (z == 0) ? tq : (z == 1) ? tk : tv;
    const size_t i = (size_t)blockIdx.x * 256 + threadIdx.x;  // (h*VEC+v)*D + d
    const int d = (int)(i % D_);
    const size_t hv = i / D_;
    const int v = (int)(hv % VEC_);
    const int h = (int)(hv / VEC_);
    T[((size_t)h * D_ + d) * VEC_ + v] = (__bf16)W[i];
}

// ---------------------------------------------------------------------------
// Kernel 1: per-head projection (bf16 in, bf16 out, f32 accumulate), async
// double-buffered LDS tiles. nounroll keeps a single accumulator register
// web (no cross-unroll phi copies). blockIdx: x = 128-row tile, y = b*H+h,
// z = 0(Q, scaled 1/8) / 1(K) / 2(V transposed Vt[bh][d][L]).
// ---------------------------------------------------------------------------
__global__ __launch_bounds__(256)
void qkv_proj_kernel(const __bf16* __restrict__ Xq,
                     const __bf16* __restrict__ Xk,
                     const __bf16* __restrict__ Xv,
                     const __bf16* __restrict__ Wtq,
                     const __bf16* __restrict__ Wtk,
                     const __bf16* __restrict__ Wtv,
                     __bf16* __restrict__ Qo,
                     __bf16* __restrict__ Ko,
                     __bf16* __restrict__ Vto)
{
    const int t    = threadIdx.x;
    const int wave = t >> 5, lane = t & 31;
    const int n    = lane & 15, half = lane >> 4;
    const int rowBase = blockIdx.x * 128;
    const int bh = blockIdx.y;
    const int b  = bh / H_, h = bh % H_;
    const int z  = blockIdx.z;

    const __bf16* Xin = (z == 0) ? Xq : (z == 1) ? Xk : Xv;
    const __bf16* Wt  = (z == 0) ? Wtq : (z == 1) ? Wtk : Wtv;

    __shared__ alignas(16) __bf16 Xs[2][128 * 32];   // [row][k]
    __shared__ alignas(16) __bf16 Wst[2][64 * 32];   // [d_out][k]

    auto load_tiles = [&](int kk, int bufi) {
        const int k0 = kk * 32;
        #pragma unroll
        for (int it = 0; it < 2; ++it) {                  // X tile 128x32
            int e = (it * 256 + t) * 8;
            int r = e >> 5, c = e & 31;
            cp16(&Xin[(size_t)(b * L_ + rowBase + r) * VEC_ + k0 + c],
                 &Xs[bufi][r * 32 + c]);
        }
        {                                                 // W slice 64x32
            int e = t * 8;
            int d = e >> 5, c = e & 31;
            cp16(&Wt[((size_t)h * D_ + d) * VEC_ + k0 + c],
                 &Wst[bufi][d * 32 + c]);
        }
    };

    v8f acc[4] = {};
    const int NK = VEC_ / 32;

    load_tiles(0, 0);
    wait_cp();
    __syncthreads();

    #pragma nounroll
    for (int kk = 0; kk < NK; ++kk) {
        const int bufi = kk & 1;
        const int knext = (kk + 1 < NK) ? kk + 1 : kk;   // clamp: branch-free
        load_tiles(knext, bufi ^ 1);                      // prefetch

        const __bf16* Xb = &Xs[bufi][0];
        const __bf16* Wb = &Wst[bufi][0];
        const v16bf a = ld_a(&Xb[(wave * 16 + n) * 32], half);
        #pragma unroll
        for (int nt = 0; nt < 4; ++nt) {
            const v16bf bw = ld16(&Wb[(nt * 16 + n) * 32 + half * 16]);
            acc[nt] = __builtin_amdgcn_wmma_f32_16x16x32_bf16(
                false, a, false, bw, (short)0, acc[nt], false, false);
        }
        wait_cp();
        __syncthreads();
    }

    if (z == 2) {
        // transposed V store: lane's 8 accumulator rows contiguous along L
        #pragma unroll
        for (int nt = 0; nt < 4; ++nt) {
            union { uint4 u; __bf16 e[8]; } pk;
            #pragma unroll
            for (int r = 0; r < 8; ++r) pk.e[r] = (__bf16)acc[nt][r];
            const int d = nt * 16 + n;
            const size_t off = (size_t)(bh * D_ + d) * L_ + rowBase + wave * 16 + 8 * half;
            *(uint4*)&Vto[off] = pk.u;
        }
    } else {
        __bf16* Out = (z == 0) ? Qo : Ko;
        const float scale = (z == 0) ? 0.125f : 1.0f;     // fold 1/sqrt(64) into Q
        #pragma unroll
        for (int nt = 0; nt < 4; ++nt)
            #pragma unroll
            for (int r = 0; r < 8; ++r) {
                int row = rowBase + wave * 16 + r + 8 * half;
                Out[(size_t)(bh * L_ + row) * D_ + nt * 16 + n] =
                    (__bf16)(acc[nt][r] * scale);
            }
    }
}

// ---------------------------------------------------------------------------
// Kernel 2: fused attention, online softmax over the FULL row (reference
// applies tril/vmask AFTER softmax: denominator = all keys, numerator masked).
// One (b,h) x 128 query rows per block; async double-buffered K/V tiles.
// ---------------------------------------------------------------------------
__global__ __launch_bounds__(256)
void flash_attn_kernel(const __bf16* __restrict__ Qws,
                       const __bf16* __restrict__ Kws,
                       const __bf16* __restrict__ Vtws,
                       const float* __restrict__ qmask,
                       const float* __restrict__ vmask,
                       float* __restrict__ Xout)
{
    const int t    = threadIdx.x;
    const int wave = t >> 5, lane = t & 31;
    const int n    = lane & 15, half = lane >> 4;
    const int rowBase = blockIdx.x * 128;
    const int bh = blockIdx.y;
    const int b  = bh / H_, h = bh % H_;

    __shared__ alignas(16) __bf16 Ks[2][32 * 64];    // [kv][d]
    __shared__ alignas(16) __bf16 Vst[2][64 * 32];   // [d][kv] (transposed Vt)
    __shared__ alignas(16) __bf16 Ps[8][16 * 32];    // per-wave P transpose

    auto load_kv = [&](int j, int bufi) {
        const int e = t * 8;
        { int r = e >> 6, c = e & 63;
          cp16(&Kws[(size_t)(bh * L_ + j * 32 + r) * D_ + c], &Ks[bufi][r * 64 + c]); }
        { int d = e >> 5, c = e & 31;
          cp16(&Vtws[(size_t)(bh * D_ + d) * L_ + j * 32 + c], &Vst[bufi][d * 32 + c]); }
    };

    // Q strip (16 rows x 64) as two A fragments held in VGPRs
    const size_t qbase = (size_t)(bh * L_ + rowBase + wave * 16 + n) * D_;
    const v16bf aq0 = ld_a(&Qws[qbase], half);
    const v16bf aq1 = ld_a(&Qws[qbase + 32], half);

    float mrun[8], lrun[8];
    #pragma unroll
    for (int r = 0; r < 8; ++r) { mrun[r] = NEG_BIG; lrun[r] = 0.0f; }
    v8f o[4] = {};

    const int qstrip_max = rowBase + wave * 16 + 15;
    const int NJ = L_ / 32;

    load_kv(0, 0);
    wait_cp();
    __syncthreads();

    #pragma nounroll
    for (int j = 0; j < NJ; ++j) {
        const int bufi = j & 1;
        const int jnext = (j + 1 < NJ) ? j + 1 : j;      // clamp: branch-free
        load_kv(jnext, bufi ^ 1);                         // prefetch next tile

        const __bf16* Kb = &Ks[bufi][0];
        const __bf16* Vb = &Vst[bufi][0];

        // S = Q @ K^T : two 16x16 n-tiles, head dim 64 = 2 wmma k-steps each
        v8f s0 = {}, s1 = {};
        #pragma unroll
        for (int kd = 0; kd < 2; ++kd) {
            const v16bf bk0 = ld16(&Kb[(     n) * 64 + kd * 32 + half * 16]);
            const v16bf bk1 = ld16(&Kb[(16 + n) * 64 + kd * 32 + half * 16]);
            const v16bf a = kd ? aq1 : aq0;
            s0 = __builtin_amdgcn_wmma_f32_16x16x32_bf16(false, a, false, bk0, (short)0, s0, false, false);
            s1 = __builtin_amdgcn_wmma_f32_16x16x32_bf16(false, a, false, bk1, (short)0, s1, false, false);
        }

        const bool doPV = (j * 32 <= qstrip_max);   // anything at/below diagonal?
        float vmA = 0.0f, vmB = 0.0f;
        if (doPV) {
            vmA = vmask[(size_t)b * L_ + j * 32 + n];
            vmB = vmask[(size_t)b * L_ + j * 32 + 16 + n];
        }

        #pragma unroll
        for (int r = 0; r < 8; ++r) {
            float v0 = s0[r], v1 = s1[r];
            float rowmax = rmax16(fmaxf(v0, v1));
            float mnew   = fmaxf(mrun[r], rowmax);
            float sc     = __expf(mrun[r] - mnew);
            float p0     = __expf(v0 - mnew);
            float p1     = __expf(v1 - mnew);
            float rowsum = rsum16(p0 + p1);          // UNMASKED: full softmax denom
            lrun[r] = lrun[r] * sc + rowsum;
            mrun[r] = mnew;
            #pragma unroll
            for (int dt = 0; dt < 4; ++dt) o[dt][r] *= sc;

            if (doPV) {
                int qg  = rowBase + wave * 16 + r + 8 * half;
                int kv0 = j * 32 + n;
                int kv1 = j * 32 + 16 + n;
                float pm0 = (kv0 <= qg) ? p0 * vmA : 0.0f;   // tril + vmask
                float pm1 = (kv1 <= qg) ? p1 * vmB : 0.0f;
                int m = r + 8 * half;
                Ps[wave][m * 32 +      n] = (__bf16)pm0;
                Ps[wave][m * 32 + 16 + n] = (__bf16)pm1;
            }
        }

        if (doPV) {
            // per-wave LDS C->A transpose; make the in-order LDS dependency
            // explicit with the CDNA5 split DS counter wait.
            asm volatile("s_wait_dscnt 0x0" ::: "memory");
            const v16bf ap = ld_a(&Ps[wave][n * 32], half);   // lane%16 = M
            #pragma unroll
            for (int dt = 0; dt < 4; ++dt) {
                const v16bf bv = ld16(&Vb[(dt * 16 + n) * 32 + half * 16]);
                o[dt] = __builtin_amdgcn_wmma_f32_16x16x32_bf16(
                    false, ap, false, bv, (short)0, o[dt], false, false);
            }
        }

        wait_cp();        // prefetched tile landed in LDS
        __syncthreads();  // everyone done reading current tile
    }

    // normalize by full-row denom, apply qmask, store concat-head layout
    #pragma unroll
    for (int r = 0; r < 8; ++r) {
        int qg = rowBase + wave * 16 + r + 8 * half;
        float s = (1.0f / lrun[r]) * qmask[(size_t)b * L_ + qg];
        #pragma unroll
        for (int dt = 0; dt < 4; ++dt)
            Xout[(size_t)(b * L_ + qg) * VEC_ + h * D_ + dt * 16 + n] = o[dt][r] * s;
    }
}

// ---------------------------------------------------------------------------
// Kernel 3: y = x + query; LayerNorm(eps=1e-3) with gamma/beta. 1 row/block.
// ---------------------------------------------------------------------------
__global__ __launch_bounds__(256)
void resid_ln_kernel(const float* __restrict__ X,
                     const float* __restrict__ q_in,
                     const float* __restrict__ gamma,
                     const float* __restrict__ beta,
                     float* __restrict__ out)
{
    const int row = blockIdx.x;
    const int t   = threadIdx.x;
    const size_t base = (size_t)row * VEC_;
    float y0 = X[base + t]       + q_in[base + t];
    float y1 = X[base + 256 + t] + q_in[base + 256 + t];

    __shared__ float red[256];
    red[t] = y0 + y1;
    __syncthreads();
    for (int s = 128; s > 0; s >>= 1) {
        if (t < s) red[t] += red[t + s];
        __syncthreads();
    }
    const float mean = red[0] * (1.0f / VEC_);
    __syncthreads();
    float d0 = y0 - mean, d1 = y1 - mean;
    red[t] = d0 * d0 + d1 * d1;
    __syncthreads();
    for (int s = 128; s > 0; s >>= 1) {
        if (t < s) red[t] += red[t + s];
        __syncthreads();
    }
    const float rstd = rsqrtf(red[0] * (1.0f / VEC_) + EPS_);
    out[base + t]       = d0 * rstd * gamma[t]       + beta[t];
    out[base + 256 + t] = d1 * rstd * gamma[256 + t] + beta[256 + t];
}

// ---------------------------------------------------------------------------
extern "C" void kernel_launch(void* const* d_in, const int* in_sizes, int n_in,
                              void* d_out, int out_size, void* d_ws, size_t ws_size,
                              hipStream_t stream) {
    const float* query = (const float*)d_in[0];
    const float* key   = (const float*)d_in[1];
    const float* value = (const float*)d_in[2];
    const float* qmask = (const float*)d_in[3];
    const float* vmask = (const float*)d_in[4];
    const float* Wq    = (const float*)d_in[5];
    const float* Wk    = (const float*)d_in[6];
    const float* Wv    = (const float*)d_in[7];
    const float* gamma = (const float*)d_in[8];
    const float* beta  = (const float*)d_in[9];
    float* out = (float*)d_out;

    const size_t NX = (size_t)B_ * L_ * VEC_;          // activation elems
    const size_t NW = (size_t)H_ * VEC_ * D_;          // weight elems per tensor
    const size_t NQ = (size_t)B_ * H_ * L_ * D_;       // Q/K/V elems per tensor

    char* p = (char*)d_ws;
    __bf16* Xqb = (__bf16*)p; p += NX * 2;
    __bf16* Xkb = (__bf16*)p; p += NX * 2;
    __bf16* Xvb = (__bf16*)p; p += NX * 2;
    __bf16* Wtq = (__bf16*)p; p += NW * 2;
    __bf16* Wtk = (__bf16*)p; p += NW * 2;
    __bf16* Wtv = (__bf16*)p; p += NW * 2;
    __bf16* Qws = (__bf16*)p; p += NQ * 2;
    __bf16* Kws = (__bf16*)p; p += NQ * 2;
    __bf16* Vtw = (__bf16*)p; p += NQ * 2;
    float*  Xb  = (float*)p;                            // ~65.5 MB total

    cvt_bf16_kernel<<<dim3((unsigned)(NX / (256 * 8)), 3), 256, 0, stream>>>(
        query, key, value, Xqb, Xkb, Xvb);
    wt_kernel<<<dim3((unsigned)(NW / 256), 3), 256, 0, stream>>>(
        Wq, Wk, Wv, Wtq, Wtk, Wtv);
    qkv_proj_kernel<<<dim3(L_ / 128, B_ * H_, 3), 256, 0, stream>>>(
        Xqb, Xkb, Xvb, Wtq, Wtk, Wtv, Qws, Kws, Vtw);
    flash_attn_kernel<<<dim3(L_ / 128, B_ * H_), 256, 0, stream>>>(
        Qws, Kws, Vtw, qmask, vmask, Xb);
    resid_ln_kernel<<<B_ * L_, 256, 0, stream>>>(Xb, query, gamma, beta, out);
}
